// LargeScaleSourceIntegration_38457137168681
// MI455X (gfx1250) — matrix operand order repaired and checked by reading, not compiled, not run on hardware
//
#include <hip/hip_runtime.h>
#include <hip/hip_bf16.h>

// ---------------------------------------------------------------------------
// Problem constants (from reference)
// ---------------------------------------------------------------------------
#define BB   2048
#define DD   512
#define HH   512
#define EE   32
#define CHH  32
#define TOPK 4

typedef __attribute__((ext_vector_type(16))) __bf16        v16bf;
typedef __attribute__((ext_vector_type(8)))  float         v8f;
typedef __attribute__((ext_vector_type(4)))  unsigned int  u32x4;

typedef __attribute__((address_space(3))) __bf16 lds_bf16_t;

union FragBF {
    u32x4 u[2];
    v16bf v;
};

// ---------------------------------------------------------------------------
// Inner WMMA K-loop: acc += A(16xK) * B(Kx16), bf16 inputs, f32 acc.
// arow: per-lane row pointer of A (row-major, K contiguous), lane row = lane&15
// brow: per-lane row pointer of B^T (row-major [N][K]),      lane col = lane&15
// half: lane>=16
// ---------------------------------------------------------------------------
static __device__ __forceinline__
v8f wmma_loop_bf16(const __bf16* __restrict__ arow,
                   const __bf16* __restrict__ brow,
                   int K, int half, v8f acc)
{
    for (int k0 = 0; k0 < K; k0 += 32) {
        FragBF a, b;
        a.u[0] = *(const u32x4*)(arow + k0 + half * 8);
        a.u[1] = *(const u32x4*)(arow + k0 + 16 + half * 8);
        b.u[0] = *(const u32x4*)(brow + k0 + half * 8);
        b.u[1] = *(const u32x4*)(brow + k0 + 16 + half * 8);
        acc = __builtin_amdgcn_wmma_f32_16x16x32_bf16(
                  false, a.v, false, b.v, (short)0, acc, false, false);
    }
    return acc;
}

// ---------------------------------------------------------------------------
// Elementwise fp32 -> bf16
// ---------------------------------------------------------------------------
__global__ void k_cvt_bf16(const float* __restrict__ src, __bf16* __restrict__ dst, int n)
{
    int i = blockIdx.x * blockDim.x + threadIdx.x;
    if (i < n) dst[i] = (__bf16)src[i];
}

// ---------------------------------------------------------------------------
// Tiled transpose + convert: src fp32 [mats][R][C] -> dst bf16 [mats][C][R]
// block (32,8), grid (C/32, R/32, mats)
// ---------------------------------------------------------------------------
__global__ void k_transpose_cvt(const float* __restrict__ src, __bf16* __restrict__ dst,
                                int R, int C)
{
    __shared__ float tile[32][33];
    int mat = blockIdx.z;
    const float* s = src + (size_t)mat * R * C;
    __bf16*      d = dst + (size_t)mat * R * C;
    int c0 = blockIdx.x * 32, r0 = blockIdx.y * 32;
    int tx = threadIdx.x, ty = threadIdx.y;
    for (int j = 0; j < 32; j += 8) {
        int r = r0 + ty + j, c = c0 + tx;
        if (r < R && c < C) tile[ty + j][tx] = s[(size_t)r * C + c];
    }
    __syncthreads();
    for (int j = 0; j < 32; j += 8) {
        int c = c0 + ty + j, r = r0 + tx;
        if (r < R && c < C) d[(size_t)c * R + r] = (__bf16)tile[tx][ty + j];
    }
}

// ---------------------------------------------------------------------------
// Zero fp32 buffer
// ---------------------------------------------------------------------------
__global__ void k_zero(float* __restrict__ p, int n)
{
    int i = blockIdx.x * blockDim.x + threadIdx.x;
    if (i < n) p[i] = 0.0f;
}

// ---------------------------------------------------------------------------
// Generic WMMA GEMM: C[M,N] = act(A[M,K] * B^T[N,K] + bias[N])
// block 256 (8 waves) computes 64(M) x 32(N); grid (M/64, N/32)
// ---------------------------------------------------------------------------
template<bool RELU, bool OUT_BF16>
__global__ __launch_bounds__(256)
void k_gemm(const __bf16* __restrict__ A, const __bf16* __restrict__ BT,
            const float* __restrict__ bias, void* __restrict__ Cout,
            int M, int N, int K)
{
    int wave = threadIdx.x >> 5;
    int lane = threadIdx.x & 31;
    int half = lane >> 4;
    int l15  = lane & 15;
    int m0 = blockIdx.x * 64 + (wave >> 1) * 16;
    int n0 = blockIdx.y * 32 + (wave & 1) * 16;

    const __bf16* arow = A  + (size_t)(m0 + l15) * K;
    const __bf16* brow = BT + (size_t)(n0 + l15) * K;

    v8f acc = {0.f, 0.f, 0.f, 0.f, 0.f, 0.f, 0.f, 0.f};
    acc = wmma_loop_bf16(arow, brow, K, half, acc);

    int   nCol = n0 + l15;
    float bs   = bias[nCol];
    for (int r = 0; r < 8; ++r) {
        int   mr = m0 + r + half * 8;
        float v  = acc[r] + bs;
        if (RELU) v = fmaxf(v, 0.f);
        if (OUT_BF16) ((__bf16*)Cout)[(size_t)mr * N + nCol] = (__bf16)v;
        else          ((float*)Cout)[(size_t)mr * N + nCol]  = v;
    }
}

// ---------------------------------------------------------------------------
// Gate softmax + top-4 selection. One wave per token (lane == expert, E==32).
// gsel[t,e] = softmax(logits)[t,e] if e in top-4 else 0.
// block 256 = 8 tokens; grid B/8
// ---------------------------------------------------------------------------
__global__ __launch_bounds__(256)
void k_gate_topk(const float* __restrict__ logits, float* __restrict__ gsel)
{
    int lane = threadIdx.x & 31;
    int wv   = threadIdx.x >> 5;
    int t    = blockIdx.x * 8 + wv;

    float l = logits[(size_t)t * EE + lane];
    float m = l;
    for (int o = 16; o; o >>= 1) m = fmaxf(m, __shfl_xor(m, o));
    float p = __expf(l - m);
    float s = p;
    for (int o = 16; o; o >>= 1) s += __shfl_xor(s, o);
    float w = p / s;

    bool sel = false;
    for (int it = 0; it < TOPK; ++it) {
        float a  = sel ? -3.4e38f : w;
        float mx = a;
        for (int o = 16; o; o >>= 1) mx = fmaxf(mx, __shfl_xor(mx, o));
        unsigned long long b = __ballot(!sel && (w == mx));
        int leader = __ffsll((long long)b) - 1;
        if (lane == leader) sel = true;
    }
    gsel[(size_t)t * EE + lane] = sel ? w : 0.f;
}

// ---------------------------------------------------------------------------
// Fused expert kernel. grid (B/16, E), block 256 (8 waves).
// Per block: 16 tokens x one expert. Early-exit if no token routed here.
//   X tile staged into LDS via CDNA5 async direct-to-LDS loads (ASYNCcnt).
//   H = relu(X*W1+b1)   -> LDS bf16   (WMMA bf16)
//   O = H*W2+b2         -> LDS bf16   (WMMA bf16)
//   c = sigmoid(relu(O*Wc1+bc1)*Wc2+bc2)
//   num += (gsel*c)*O ; den += gsel*c   (global fp32 atomics, masked)
// ---------------------------------------------------------------------------
__global__ __launch_bounds__(256)
void k_expert(const __bf16* __restrict__ xb,
              const __bf16* __restrict__ W1T, const float* __restrict__ b1,
              const __bf16* __restrict__ W2T, const float* __restrict__ b2,
              const __bf16* __restrict__ Wc1T, const float* __restrict__ bc1,
              const float* __restrict__ Wc2,  const float* __restrict__ bc2,
              const float* __restrict__ gsel,
              float* __restrict__ num, float* __restrict__ den)
{
    __shared__ __align__(16) __bf16 sX[16][DD];
    __shared__ __align__(16) __bf16 sH[16][HH];
    __shared__ __align__(16) __bf16 sO[16][DD];
    __shared__ float sC1[16][CHH];
    __shared__ float sCw[16];
    __shared__ int   sAny;

    int e   = blockIdx.y;
    int m0  = blockIdx.x * 16;
    int tid = threadIdx.x;

    if (tid == 0) sAny = 0;
    __syncthreads();
    if (tid < 16) {
        float g = gsel[(size_t)(m0 + tid) * EE + e];
        sCw[tid] = g;
        if (g != 0.f) atomicOr(&sAny, 1);
    }
    __syncthreads();
    if (!sAny) return;  // no token in this tile routed to expert e

    // ---- X-tile staging (16 x 512 bf16): async global -> LDS, no VGPR hop ----
    {
        int row = tid >> 4;
        int cc  = (tid & 15) * 32;                       // 32 bf16 = 64B per thread
        const __bf16* gp = xb + (size_t)(m0 + row) * DD + cc;
        // LDS byte offset of destination (generic -> AS3 -> int)
        unsigned int lds_off =
            (unsigned int)(size_t)(lds_bf16_t*)(&sX[row][cc]);
        unsigned long long gaddr = (unsigned long long)(size_t)gp;
        // 4 x 16B async copies per thread (GLOBAL_LOAD_ASYNC_TO_LDS_B128)
        for (int i = 0; i < 4; ++i) {
            asm volatile("global_load_async_to_lds_b128 %0, %1, off"
                         :: "v"(lds_off + i * 16u), "v"(gaddr + i * 16ull)
                         : "memory");
        }
        asm volatile("s_wait_asynccnt 0" ::: "memory");
    }
    __syncthreads();

    int wave = tid >> 5, lane = tid & 31, half = lane >> 4, l15 = lane & 31 & 15;
    l15 = lane & 15;

    // ---- H = relu(X*W1 + b1) ----  32 N-tiles, 4 per wave
    for (int nt = wave; nt < HH / 16; nt += 8) {
        int n = nt * 16 + l15;
        // warm next tile's B row into cache while this tile computes
        if (nt + 8 < HH / 16)
            __builtin_prefetch(W1T + ((size_t)e * HH + n + 128) * DD, 0, 1);
        v8f acc = {0.f, 0.f, 0.f, 0.f, 0.f, 0.f, 0.f, 0.f};
        acc = wmma_loop_bf16(&sX[l15][0], W1T + ((size_t)e * HH + n) * DD, DD, half, acc);
        float bs = b1[e * HH + n];
        for (int r = 0; r < 8; ++r) {
            int M = r + half * 8;
            sH[M][n] = (__bf16)fmaxf(acc[r] + bs, 0.f);
        }
    }
    __syncthreads();

    // ---- O = H*W2 + b2 ----
    for (int nt = wave; nt < DD / 16; nt += 8) {
        int n = nt * 16 + l15;
        if (nt + 8 < DD / 16)
            __builtin_prefetch(W2T + ((size_t)e * DD + n + 128) * HH, 0, 1);
        v8f acc = {0.f, 0.f, 0.f, 0.f, 0.f, 0.f, 0.f, 0.f};
        acc = wmma_loop_bf16(&sH[l15][0], W2T + ((size_t)e * DD + n) * HH, HH, half, acc);
        float bs = b2[e * DD + n];
        for (int r = 0; r < 8; ++r) {
            int M = r + half * 8;
            sO[M][n] = (__bf16)(acc[r] + bs);
        }
    }
    __syncthreads();

    // ---- conf1 = relu(O*Wc1 + bc1)  [16,32] ----  2 N-tiles on waves 0..1
    if (wave < 2) {
        int n = wave * 16 + l15;
        v8f acc = {0.f, 0.f, 0.f, 0.f, 0.f, 0.f, 0.f, 0.f};
        acc = wmma_loop_bf16(&sO[l15][0], Wc1T + ((size_t)e * CHH + n) * DD, DD, half, acc);
        float bs = bc1[e * CHH + n];
        for (int r = 0; r < 8; ++r) {
            int M = r + half * 8;
            sC1[M][n] = fmaxf(acc[r] + bs, 0.f);
        }
    }
    __syncthreads();

    // ---- conf2 + sigmoid, fold into cw ----
    if (tid < 16) {
        float s = bc2[e];
        for (int ch = 0; ch < CHH; ++ch) s += sC1[tid][ch] * Wc2[e * CHH + ch];
        float conf = 1.f / (1.f + __expf(-s));
        sCw[tid] *= conf;
    }
    __syncthreads();

    // ---- masked scatter: num += cw*O, den += cw ----
    {
        int   m     = tid >> 4;
        int   dbase = (tid & 15) * 32;
        float cw    = sCw[m];
        if (cw != 0.f) {
            float* np = num + (size_t)(m0 + m) * DD + dbase;
            for (int i = 0; i < 32; ++i)
                atomicAdd(np + i, cw * (float)sO[m][dbase + i]);
            if ((tid & 15) == 0) atomicAdd(&den[m0 + m], cw);
        }
    }
}

// ---------------------------------------------------------------------------
// out = num / (den + eps)
// ---------------------------------------------------------------------------
__global__ void k_finalize(const float* __restrict__ num, const float* __restrict__ den,
                           float* __restrict__ out, int n)
{
    int i = blockIdx.x * blockDim.x + threadIdx.x;
    if (i < n) out[i] = num[i] / (den[i >> 9] + 1e-6f);  // D = 512
}

// ---------------------------------------------------------------------------
// Host launcher
// ---------------------------------------------------------------------------
extern "C" void kernel_launch(void* const* d_in, const int* in_sizes, int n_in,
                              void* d_out, int out_size, void* d_ws, size_t ws_size,
                              hipStream_t stream)
{
    (void)in_sizes; (void)n_in; (void)out_size; (void)ws_size;

    const float* x   = (const float*)d_in[0];
    const float* W1  = (const float*)d_in[1];
    const float* b1  = (const float*)d_in[2];
    const float* W2  = (const float*)d_in[3];
    const float* b2  = (const float*)d_in[4];
    const float* Wg1 = (const float*)d_in[5];
    const float* bg1 = (const float*)d_in[6];
    const float* Wg2 = (const float*)d_in[7];
    const float* bg2 = (const float*)d_in[8];
    const float* Wc1 = (const float*)d_in[9];
    const float* bc1 = (const float*)d_in[10];
    const float* Wc2 = (const float*)d_in[11];
    const float* bc2 = (const float*)d_in[12];
    float* out = (float*)d_out;

    // workspace carve-out (256B aligned slabs)
    char*  base = (char*)d_ws;
    size_t off  = 0;
    auto alloc = [&](size_t bytes) -> char* {
        off = (off + 255) & ~(size_t)255;
        char* p = base + off;
        off += bytes;
        return p;
    };
    __bf16* xb    = (__bf16*)alloc((size_t)BB * DD * 2);
    __bf16* W1T   = (__bf16*)alloc((size_t)EE * DD * HH * 2);
    __bf16* W2T   = (__bf16*)alloc((size_t)EE * HH * DD * 2);
    __bf16* Wc1T  = (__bf16*)alloc((size_t)EE * DD * CHH * 2);
    __bf16* Wg1T  = (__bf16*)alloc((size_t)DD * HH * 2);
    __bf16* Wg2T  = (__bf16*)alloc((size_t)HH * EE * 2);
    __bf16* Hg    = (__bf16*)alloc((size_t)BB * HH * 2);
    float*  logit = (float*) alloc((size_t)BB * EE * 4);
    float*  gsel  = (float*) alloc((size_t)BB * EE * 4);
    float*  num   = (float*) alloc((size_t)BB * DD * 4);
    float*  den   = (float*) alloc((size_t)BB * 4);

    dim3 tb(32, 8);

    // 1) precision/layout prep
    k_cvt_bf16<<<(BB * DD) / 256, 256, 0, stream>>>(x, xb, BB * DD);
    k_transpose_cvt<<<dim3(HH / 32, DD / 32, EE), tb, 0, stream>>>(W1,  W1T,  DD, HH);
    k_transpose_cvt<<<dim3(DD / 32, HH / 32, EE), tb, 0, stream>>>(W2,  W2T,  HH, DD);
    k_transpose_cvt<<<dim3(CHH / 32, DD / 32, EE), tb, 0, stream>>>(Wc1, Wc1T, DD, CHH);
    k_transpose_cvt<<<dim3(HH / 32, DD / 32, 1), tb, 0, stream>>>(Wg1, Wg1T, DD, HH);
    k_transpose_cvt<<<dim3(EE / 32, HH / 32, 1), tb, 0, stream>>>(Wg2, Wg2T, HH, EE);

    // 2) zero accumulators (fresh every call: graph-replay safe)
    k_zero<<<(BB * DD + 255) / 256, 256, 0, stream>>>(num, BB * DD);
    k_zero<<<(BB + 255) / 256, 256, 0, stream>>>(den, BB);

    // 3) gate
    k_gemm<true, true><<<dim3(BB / 64, HH / 32), 256, 0, stream>>>(xb, Wg1T, bg1, Hg, BB, HH, DD);
    k_gemm<false, false><<<dim3(BB / 64, EE / 32), 256, 0, stream>>>(Hg, Wg2T, bg2, logit, BB, EE, HH);
    k_gate_topk<<<BB / 8, 256, 0, stream>>>(logit, gsel);

    // 4) experts (fused MLP + confidence + masked scatter)
    k_expert<<<dim3(BB / 16, EE), 256, 0, stream>>>(xb, W1T, b1, W2T, b2,
                                                    Wc1T, bc1, Wc2, bc2,
                                                    gsel, num, den);

    // 5) combine
    k_finalize<<<(BB * DD) / 256, 256, 0, stream>>>(num, den, out, BB * DD);
}
